// GAT_6820408066447
// MI455X (gfx1250) — matrix operand order repaired and checked by reading, not compiled, not run on hardware
//
#include <hip/hip_runtime.h>

typedef __attribute__((ext_vector_type(16))) __bf16 v16bf;
typedef __attribute__((ext_vector_type(8)))  float  v8f;

#define NNODE 2048
#define DF    128
#define NBATCH 4
#define TOTROWS (NBATCH * NNODE)   // 8192
#define TSTRIDE 40                 // bf16 elems per f-row in LDS B-tile (80 B, bank-padded)

// ---------- helpers ----------
__device__ __forceinline__ __bf16 f2bf(float f) {
    union { float f; unsigned u; } c; c.f = f;
    unsigned r = c.u + 0x7FFFu + ((c.u >> 16) & 1u);   // RNE
    union { unsigned short s; __bf16 h; } o; o.s = (unsigned short)(r >> 16);
    return o.h;
}

union BPack  { unsigned int u;  __bf16 h[2]; };
union BPack4 { uint4 u4;        __bf16 h[8]; };
union CPack4 { uint2 u2;        __bf16 h[4]; };

// ---------- f32 -> bf16 conversion (4 elems / thread) ----------
__global__ __launch_bounds__(256) void cvt_bf16_kernel(
    const float* __restrict__ in, __bf16* __restrict__ out, int n4)
{
    int i = blockIdx.x * blockDim.x + threadIdx.x;
    if (i >= n4) return;
    float4 v = ((const float4*)in)[i];
    CPack4 o;
    o.h[0] = f2bf(v.x); o.h[1] = f2bf(v.y); o.h[2] = f2bf(v.z); o.h[3] = f2bf(v.w);
    ((uint2*)out)[i] = o.u2;
}

// ---------- adjacency mask bitset: bit(i,j) = adj[i,j]>0 && j!=i ----------
__global__ __launch_bounds__(256) void mask_kernel(
    const float* __restrict__ adj, unsigned int* __restrict__ umask)
{
    int t = blockIdx.x * blockDim.x + threadIdx.x;   // NNODE*64 threads
    int i = t >> 6, w = t & 63;
    const float* row = adj + (size_t)i * NNODE + w * 32;
    unsigned int m = 0;
#pragma unroll
    for (int b2 = 0; b2 < 32; ++b2) {
        int j = w * 32 + b2;
        if (row[b2] > 0.f && j != i) m |= (1u << b2);
    }
    umask[t] = m;
}

// ---------- linear layer: H[8192,128] = X_bf16 @ W^T + b  (WMMA bf16) ----------
__global__ __launch_bounds__(256) void gemm_kernel(
    const __bf16* __restrict__ X, const __bf16* __restrict__ W,
    const float*  __restrict__ bias,
    float* __restrict__ H, __bf16* __restrict__ HT, int storeT)
{
    __shared__ __align__(16) __bf16 sW[DF * DF];     // 32 KB
    {
        const uint4* src = (const uint4*)W;
        uint4* dst = (uint4*)sW;
        for (int v = threadIdx.x; v < DF * DF / 8; v += blockDim.x) dst[v] = src[v];
    }
    __syncthreads();

    int wave = threadIdx.x >> 5, lane = threadIdx.x & 31;
    int mloc = lane & 15, g = lane >> 4;
    int rowbase = (blockIdx.x * 8 + wave) * 16;
    const __bf16* Xrow = X + (size_t)(rowbase + mloc) * DF;

    v8f acc[8];
#pragma unroll
    for (int t = 0; t < 8; ++t)
#pragma unroll
        for (int r = 0; r < 8; ++r) acc[t][r] = 0.f;

#pragma unroll
    for (int kk = 0; kk < 4; ++kk) {
        int k0 = kk * 32;
        v16bf a;
#pragma unroll
        for (int v = 0; v < 8; ++v) {          // A frag: row m=L&15, k=16g+s contiguous
            int kb = 16 * g + 2 * v;
            BPack p; p.u = *(const unsigned int*)(Xrow + k0 + kb);
            a[2 * v]     = p.h[0];
            a[2 * v + 1] = p.h[1];
        }
#pragma unroll
        for (int t = 0; t < 8; ++t) {          // B frag: col n=L&15, k=16g+s contiguous
            int f = t * 16 + mloc;
            const uint4* bp = (const uint4*)(sW + (size_t)f * DF + k0 + 16 * g);
            BPack4 q0, q1; q0.u4 = bp[0]; q1.u4 = bp[1];
            v16bf bb;
#pragma unroll
            for (int e = 0; e < 8; ++e) { bb[e] = q0.h[e]; bb[8 + e] = q1.h[e]; }
            acc[t] = __builtin_amdgcn_wmma_f32_16x16x32_bf16(
                false, a, false, bb, (short)0, acc[t], false, false);
        }
    }

#pragma unroll
    for (int t = 0; t < 8; ++t) {
        int fcol = t * 16 + mloc;
        float bv = bias[fcol];
#pragma unroll
        for (int r = 0; r < 8; ++r) {          // C/D: row r+8g, col L&15
            int row = rowbase + r + 8 * g;
            float v = acc[t][r] + bv;
            H[(size_t)row * DF + fcol] = v;
            if (storeT) {
                int b = row >> 11, i = row & (NNODE - 1);
                HT[((size_t)b * DF + fcol) * NNODE + i] = f2bf(v);
            }
        }
    }
}

// ---------- s = h@a_src, d = h@a_dst (per row) ----------
__global__ __launch_bounds__(256) void sd_kernel(
    const float* __restrict__ H, const float* __restrict__ asrc,
    const float* __restrict__ adst, float* __restrict__ S, float* __restrict__ D)
{
    int row = blockIdx.x * blockDim.x + threadIdx.x;   // 8192
    const float4* h4 = (const float4*)(H + (size_t)row * DF);
    const float4* a4 = (const float4*)asrc;
    const float4* d4 = (const float4*)adst;
    float s = 0.f, d = 0.f;
#pragma unroll
    for (int q = 0; q < DF / 4; ++q) {
        float4 h = h4[q], a = a4[q], dd = d4[q];
        s += h.x * a.x + h.y * a.y + h.z * a.z + h.w * a.w;
        d += h.x * dd.x + h.y * dd.y + h.z * dd.z + h.w * dd.w;
    }
    S[row] = s; D[row] = d;
}

// ---------- softmax pass 1: row max & sum over masked relu(s_i+d_j+ab) ----------
__global__ __launch_bounds__(256) void pass1_kernel(
    const float* __restrict__ S, const float* __restrict__ D,
    const unsigned int* __restrict__ umask, const float* __restrict__ ab_ptr,
    float* __restrict__ Mx, float* __restrict__ Ll)
{
    __shared__ float dsh[NNODE];               // 8 KB: d[b,:]
    int b = blockIdx.x >> 3;                   // 8 blocks / batch
    int i = (blockIdx.x & 7) * 256 + threadIdx.x;
    const float* Db = D + (size_t)b * NNODE;
    for (int j = threadIdx.x; j < NNODE; j += 256) dsh[j] = Db[j];
    __syncthreads();

    float si = S[(size_t)b * NNODE + i] + *ab_ptr;
    const unsigned int* mrow = umask + (size_t)i * 64;

    float m = 0.f;                             // relu >= 0 and row has masked entries
    for (int w = 0; w < 64; ++w) {
        unsigned int mw = mrow[w];
#pragma unroll
        for (int bit = 0; bit < 32; ++bit)
            if (mw & (1u << bit)) {
                float r = si + dsh[w * 32 + bit];
                r = r > 0.f ? r : 0.f;
                m = fmaxf(m, r);
            }
    }
    float l = 0.f;
    for (int w = 0; w < 64; ++w) {
        unsigned int mw = mrow[w];
#pragma unroll
        for (int bit = 0; bit < 32; ++bit)
            if (mw & (1u << bit)) {
                float r = si + dsh[w * 32 + bit];
                r = r > 0.f ? r : 0.f;
                l += __expf(r - m);
            }
    }
    Mx[(size_t)b * NNODE + i] = m;
    Ll[(size_t)b * NNODE + i] = l;
}

// ---------- pass 2: out = softmax(e) @ h + h_other  [+relu+BN -> y_bf16] ----------
// flash-style: P tile (16x32 bf16) built in registers; B tiles (h^T) streamed
// into LDS with double-buffered global_load_async_to_lds_b128 (ASYNCcnt).
__global__ __launch_bounds__(256) void pass2_kernel(
    const __bf16* __restrict__ HT, const float* __restrict__ Hadd,
    const float* __restrict__ S, const float* __restrict__ D,
    const float* __restrict__ Mx, const float* __restrict__ Ll,
    const unsigned int* __restrict__ umask, const float* __restrict__ ab_ptr,
    const float* __restrict__ bn_gamma, const float* __restrict__ bn_beta,
    const float* __restrict__ bn_mean,  const float* __restrict__ bn_var,
    float* __restrict__ OutF, __bf16* __restrict__ OutB, int use_bn)
{
    __shared__ float dsh[NNODE];                               // 8 KB
    __shared__ __align__(16) __bf16 btile[2][DF * TSTRIDE];    // 2 x 10 KB

    int b = blockIdx.x >> 4;                   // 16 blocks / batch
    int tid = threadIdx.x;
    int wave = tid >> 5, lane = tid & 31;
    int mloc = lane & 15, g = lane >> 4;
    int i0 = ((blockIdx.x & 15) * 8 + wave) * 16;

    const float* Db = D + (size_t)b * NNODE;
    for (int j = tid; j < NNODE; j += 256) dsh[j] = Db[j];

    const __bf16* HTb = HT + (size_t)b * DF * NNODE;

    // async B-tile copy: 128 f-rows x 64 B = 512 b128 chunks; 2 per thread
    unsigned lds0 = (unsigned)(size_t)(void*)&btile[0][0];
    unsigned lds1 = (unsigned)(size_t)(void*)&btile[1][0];
    int f0 = tid >> 2, c = tid & 3;
    int f1 = 64 + (tid >> 2);
    auto issue_tile = [&](unsigned ldsbase, int j0) {
        unsigned long long g0 =
            (unsigned long long)(size_t)(HTb + (size_t)f0 * NNODE + j0 + c * 8);
        unsigned long long g1 =
            (unsigned long long)(size_t)(HTb + (size_t)f1 * NNODE + j0 + c * 8);
        unsigned l0 = ldsbase + (unsigned)(f0 * TSTRIDE * 2 + c * 16);
        unsigned l1 = ldsbase + (unsigned)(f1 * TSTRIDE * 2 + c * 16);
        asm volatile("global_load_async_to_lds_b128 %0, %1, off"
                     :: "v"(l0), "v"(g0) : "memory");
        asm volatile("global_load_async_to_lds_b128 %0, %1, off"
                     :: "v"(l1), "v"(g1) : "memory");
    };

    int irow = i0 + mloc;
    float si = S[(size_t)b * NNODE + irow] + *ab_ptr;
    float mi = Mx[(size_t)b * NNODE + irow];
    const unsigned int* mrow = umask + (size_t)irow * 64;

    v8f acc[8];
#pragma unroll
    for (int t = 0; t < 8; ++t)
#pragma unroll
        for (int r = 0; r < 8; ++r) acc[t][r] = 0.f;

    issue_tile(lds0, 0);                       // prefetch tile 0 (ASYNCcnt += 2)
    __syncthreads();                           // dsh visible

    for (int j0 = 0; j0 < NNODE; j0 += 32) {
        int cur = (j0 >> 5) & 1;
        bool more = (j0 + 32) < NNODE;
        if (more) {
            issue_tile(cur ? lds0 : lds1, j0 + 32);   // stream next tile
            asm volatile("s_wait_asynccnt 0x2" ::: "memory");  // cur tile landed
        } else {
            asm volatile("s_wait_asynccnt 0x0" ::: "memory");
        }
        __syncthreads();                       // tile visible to all waves

        unsigned int mw = mrow[j0 >> 5];
        v16bf a;
#pragma unroll
        for (int v = 0; v < 8; ++v) {
            int kb = 16 * g + 2 * v;
#pragma unroll
            for (int p = 0; p < 2; ++p) {
                int kk = kb + p;
                float pw = 0.f;
                if (mw & (1u << kk)) {
                    float r = si + dsh[j0 + kk];
                    r = r > 0.f ? r : 0.f;
                    pw = __expf(r - mi);       // /l deferred to epilogue (per C-row)
                }
                a[2 * v + p] = f2bf(pw);
            }
        }
        const __bf16* tb = &btile[cur][0];
#pragma unroll
        for (int t = 0; t < 8; ++t) {
            const uint4* bp =
                (const uint4*)(tb + (size_t)(t * 16 + mloc) * TSTRIDE + 16 * g);
            BPack4 q0, q1; q0.u4 = bp[0]; q1.u4 = bp[1];
            v16bf bb;
#pragma unroll
            for (int e = 0; e < 8; ++e) { bb[e] = q0.h[e]; bb[8 + e] = q1.h[e]; }
            acc[t] = __builtin_amdgcn_wmma_f32_16x16x32_bf16(
                false, a, false, bb, (short)0, acc[t], false, false);
        }
        __syncthreads();                       // all waves done before overwrite
    }

#pragma unroll
    for (int r = 0; r < 8; ++r) {
        int io = i0 + r + 8 * g;
        float invl = 1.f / Ll[(size_t)b * NNODE + io];
        float mean = 0.f, scale = 0.f, beta = 0.f;
        if (use_bn) {
            mean  = bn_mean[io];
            scale = rsqrtf(bn_var[io] + 1e-5f) * bn_gamma[io];
            beta  = bn_beta[io];
        }
#pragma unroll
        for (int t = 0; t < 8; ++t) {
            int fcol = t * 16 + mloc;
            size_t oi = ((size_t)b * NNODE + io) * DF + fcol;
            float v = acc[t][r] * invl + Hadd[oi];
            if (use_bn) {
                v = v > 0.f ? v : 0.f;
                v = (v - mean) * scale + beta;
                OutB[oi] = f2bf(v);
            } else {
                OutF[oi] = v;
            }
        }
    }
}

// ---------- host ----------
extern "C" void kernel_launch(void* const* d_in, const int* in_sizes, int n_in,
                              void* d_out, int out_size, void* d_ws, size_t ws_size,
                              hipStream_t stream) {
    (void)in_sizes; (void)n_in; (void)out_size; (void)ws_size;
    const float* x        = (const float*)d_in[0];
    const float* adj      = (const float*)d_in[1];
    const float* W1       = (const float*)d_in[2];
    const float* b1       = (const float*)d_in[3];
    const float* a_src    = (const float*)d_in[4];
    const float* a_dst    = (const float*)d_in[5];
    const float* a_b      = (const float*)d_in[6];
    const float* bn_gamma = (const float*)d_in[7];
    const float* bn_beta  = (const float*)d_in[8];
    const float* bn_mean  = (const float*)d_in[9];
    const float* bn_var   = (const float*)d_in[10];
    float* out = (float*)d_out;

    char* p = (char*)d_ws;
    auto alloc = [&](size_t bytes) -> void* {
        void* r = (void*)p; p += (bytes + 255) & ~(size_t)255; return r;
    };
    __bf16* xbf = (__bf16*)alloc((size_t)TOTROWS * DF * 2);
    __bf16* Wbf = (__bf16*)alloc((size_t)4 * DF * DF * 2);
    float*  HaF = (float*) alloc((size_t)TOTROWS * DF * 4);
    __bf16* HaT = (__bf16*)alloc((size_t)TOTROWS * DF * 2);
    float*  HbF = (float*) alloc((size_t)TOTROWS * DF * 4);
    float*  S   = (float*) alloc((size_t)TOTROWS * 4);
    float*  D   = (float*) alloc((size_t)TOTROWS * 4);
    float*  Mx  = (float*) alloc((size_t)TOTROWS * 4);
    float*  Ll  = (float*) alloc((size_t)TOTROWS * 4);
    __bf16* ybf = (__bf16*)alloc((size_t)TOTROWS * DF * 2);
    unsigned int* um = (unsigned int*)alloc((size_t)NNODE * 64 * 4);

    // conversions + mask
    cvt_bf16_kernel<<<(TOTROWS * DF / 4 + 255) / 256, 256, 0, stream>>>(x, xbf, TOTROWS * DF / 4);
    cvt_bf16_kernel<<<(4 * DF * DF / 4 + 255) / 256, 256, 0, stream>>>(W1, Wbf, 4 * DF * DF / 4);
    mask_kernel<<<NNODE * 64 / 256, 256, 0, stream>>>(adj, um);

    // ---- block 1 (layers 0,1) on x ----
    gemm_kernel<<<64, 256, 0, stream>>>(xbf, Wbf + 0 * DF * DF, b1 + 0 * DF, HaF, HaT, 1);
    gemm_kernel<<<64, 256, 0, stream>>>(xbf, Wbf + 1 * DF * DF, b1 + 1 * DF, HbF, nullptr, 0);
    sd_kernel<<<TOTROWS / 256, 256, 0, stream>>>(HaF, a_src + 0 * DF, a_dst + 0 * DF, S, D);
    pass1_kernel<<<TOTROWS / 256, 256, 0, stream>>>(S, D, um, a_b + 0, Mx, Ll);
    pass2_kernel<<<NBATCH * 16, 256, 0, stream>>>(HaT, HbF, S, D, Mx, Ll, um, a_b + 0,
                                                  bn_gamma, bn_beta, bn_mean, bn_var,
                                                  nullptr, ybf, 1);

    // ---- block 2 (layers 2,3) on y ----
    gemm_kernel<<<64, 256, 0, stream>>>(ybf, Wbf + 2 * DF * DF, b1 + 2 * DF, HaF, HaT, 1);
    gemm_kernel<<<64, 256, 0, stream>>>(ybf, Wbf + 3 * DF * DF, b1 + 3 * DF, HbF, nullptr, 0);
    sd_kernel<<<TOTROWS / 256, 256, 0, stream>>>(HaF, a_src + 2 * DF, a_dst + 2 * DF, S, D);
    pass1_kernel<<<TOTROWS / 256, 256, 0, stream>>>(S, D, um, a_b + 2, Mx, Ll);
    pass2_kernel<<<NBATCH * 16, 256, 0, stream>>>(HaT, HbF, S, D, Mx, Ll, um, a_b + 2,
                                                  nullptr, nullptr, nullptr, nullptr,
                                                  out, nullptr, 0);
}
